// CSA_52364241273370
// MI455X (gfx1250) — compile-verified, matrix operand order checked by reading, not run on hardware
//
#include <hip/hip_runtime.h>

// ---------------------------------------------------------------------------
// Convolutional self-attention (CSA) for MI455X / gfx1250, wave32 + WMMA bf16.
// Shapes fixed by the reference: B=4, H=W=128, C=128, HEADS=4, hd=32, K2=9.
// Pipeline:
//   K0  : cast x / w_qkv(padded 324->336) / w_v / w_proj  to bf16
//   K1  : att_raw[M,336] = xb @ wqpadT        (WMMA bf16, 16x16 tiles, N=21 odd)
//   K2  : softmax over q (groups of 9), scale 1/sqrt(32) -> attb bf16 [M,324]
//   K3  : V[p][M,128] = shift_p(xb) @ w_v[p]T (9 WMMA GEMMs, 16x32 tiles/wave)
//   K4  : attention-apply + fold gather -> yb bf16 [M,128]   (VALU; V in L2)
//   K5  : out[M,128] = yb @ w_projT           (WMMA bf16, 16x32 tiles/wave)
// All fragments are preloaded before the WMMA sequence so the load clause
// overlaps the matrix pipe (partial s_wait_loadcnt instead of wait-0 per op).
// ---------------------------------------------------------------------------

typedef __attribute__((ext_vector_type(8)))  __bf16 v8bf;
typedef __attribute__((ext_vector_type(16))) __bf16 v16bf;
typedef __attribute__((ext_vector_type(8)))  float  v8f;

#define M_TOT   65536      // B*H*W
#define C_DIM   128
#define NPAD    336        // 324 padded to 21 tiles of 16
#define NATT    324

__device__ __forceinline__ __bf16 f2bf(float f) {
  unsigned u = __builtin_bit_cast(unsigned, f);
  unsigned r = (u + 0x7fffu + ((u >> 16) & 1u)) >> 16;
  unsigned short s = (unsigned short)r;
  return __builtin_bit_cast(__bf16, s);
}
__device__ __forceinline__ float bf2f(__bf16 h) {
  unsigned short s = __builtin_bit_cast(unsigned short, h);
  unsigned u = ((unsigned)s) << 16;
  return __builtin_bit_cast(float, u);
}

// Load one 16x32 bf16 WMMA operand fragment for this lane.
// ISA layout (16-bit A / Bt, row-major [16][K] with ld=128):
//   lane L (L<16): row L, K = {0..7, 16..23} ; lane L+16: row L, K = {8..15, 24..31}
// i.e. two contiguous 8-element (16B) chunks at +0 and +16 elements.
__device__ __forceinline__ v16bf load_frag(const __bf16* p) {
  v8bf lo = *(const v8bf*)(p);
  v8bf hi = *(const v8bf*)(p + 16);
  v16bf r;
#pragma unroll
  for (int i = 0; i < 8; ++i) { r[i] = lo[i]; r[i + 8] = hi[i]; }
  return r;
}

// ---------------- K0: conversions / packing ----------------
__global__ void cvt_f32_bf16(const float* __restrict__ src, __bf16* __restrict__ dst, int n) {
  int t = blockIdx.x * blockDim.x + threadIdx.x;
  if (t < n) dst[t] = f2bf(src[t]);
}

__global__ void pack_wq(const float* __restrict__ w, __bf16* __restrict__ dst) {
  int t = blockIdx.x * blockDim.x + threadIdx.x;      // t = o*128 + k, o < 336
  if (t >= NPAD * C_DIM) return;
  int o = t >> 7;
  dst[t] = (o < NATT) ? f2bf(w[t]) : f2bf(0.0f);
}

// ---------------- K1: generic bf16 GEMM, D = A @ Bt^T (f32 out) ------------
// A: [M,128] row-major bf16;  Bt: [Nt*16,128] row-major bf16 (i.e. B^T);
// D: [M, ldn] f32. One wave per 16x16 tile, K=128 in 4 WMMA steps.
__global__ void gemm_bf16_f32(const __bf16* __restrict__ A, const __bf16* __restrict__ Bt,
                              float* __restrict__ D, int Nt, int ldn) {
  int wave = (blockIdx.x * blockDim.x + threadIdx.x) >> 5;
  int lane = threadIdx.x & 31;
  int tn = wave % Nt, tm = wave / Nt;
  int row = lane & 15, hi = lane >> 4;

  const __bf16* a = A  + (size_t)(tm * 16 + row) * C_DIM + hi * 8;
  const __bf16* b = Bt + (size_t)(tn * 16 + row) * C_DIM + hi * 8;

  v16bf af[4], bf[4];
#pragma unroll
  for (int kk = 0; kk < 4; ++kk) af[kk] = load_frag(a + kk * 32);
#pragma unroll
  for (int kk = 0; kk < 4; ++kk) bf[kk] = load_frag(b + kk * 32);

  v8f acc = {};
#pragma unroll
  for (int kk = 0; kk < 4; ++kk)
    acc = __builtin_amdgcn_wmma_f32_16x16x32_bf16(false, af[kk], false, bf[kk],
                                                  (short)0, acc, false, false);

  // D layout: lane n = lane&15, rows m = (lane>>4)*8 + r
  float* d = D + (size_t)(tm * 16 + hi * 8) * ldn + tn * 16 + row;
#pragma unroll
  for (int r = 0; r < 8; ++r) d[(size_t)r * ldn] = acc[r];
}

// ---------------- K5: bf16 GEMM, 16x32 tile per wave (f32 out, N=128) ------
// Wave computes two adjacent 16-column tiles: A frags loaded once, two
// independent accumulators keep the XDL pipe busy across WMMA hazards.
__global__ void gemm_bf16_f32_x2(const __bf16* __restrict__ A, const __bf16* __restrict__ Bt,
                                 float* __restrict__ D) {
  int wave = (blockIdx.x * blockDim.x + threadIdx.x) >> 5;
  int lane = threadIdx.x & 31;
  int tp = wave & 3;             // 128/32 = 4 column-pair tiles
  int tm = wave >> 2;
  int row = lane & 15, hi = lane >> 4;

  const __bf16* a  = A  + (size_t)(tm * 16 + row) * C_DIM + hi * 8;
  const __bf16* b0 = Bt + (size_t)(tp * 32 + row) * C_DIM + hi * 8;
  const __bf16* b1 = b0 + (size_t)16 * C_DIM;

  v16bf af[4], bf0[4], bf1[4];
#pragma unroll
  for (int kk = 0; kk < 4; ++kk) af[kk]  = load_frag(a  + kk * 32);
#pragma unroll
  for (int kk = 0; kk < 4; ++kk) bf0[kk] = load_frag(b0 + kk * 32);
#pragma unroll
  for (int kk = 0; kk < 4; ++kk) bf1[kk] = load_frag(b1 + kk * 32);

  v8f acc0 = {}, acc1 = {};
#pragma unroll
  for (int kk = 0; kk < 4; ++kk) {
    acc0 = __builtin_amdgcn_wmma_f32_16x16x32_bf16(false, af[kk], false, bf0[kk],
                                                   (short)0, acc0, false, false);
    acc1 = __builtin_amdgcn_wmma_f32_16x16x32_bf16(false, af[kk], false, bf1[kk],
                                                   (short)0, acc1, false, false);
  }

  float* d = D + (size_t)(tm * 16 + hi * 8) * C_DIM + tp * 32 + row;
#pragma unroll
  for (int r = 0; r < 8; ++r) {
    d[(size_t)r * C_DIM]      = acc0[r];
    d[(size_t)r * C_DIM + 16] = acc1[r];
  }
}

// ---------------- K2: softmax over q (9 elems), scaled ----------------
__global__ void softmax9(const float* __restrict__ att_raw, __bf16* __restrict__ attb) {
  int t = blockIdx.x * blockDim.x + threadIdx.x;   // t = bn*36 + (h*9+p)
  if (t >= M_TOT * 36) return;
  int bn = t / 36, r = t % 36;
  const float* src = att_raw + (size_t)bn * NPAD + r * 9;
  __bf16*      dst = attb    + (size_t)bn * NATT + r * 9;
  const float scale = 0.17677669529663687f;        // 32^-0.5
  float v[9], mx = -1e30f;
#pragma unroll
  for (int q = 0; q < 9; ++q) { v[q] = src[q] * scale; mx = fmaxf(mx, v[q]); }
  float s = 0.f;
#pragma unroll
  for (int q = 0; q < 9; ++q) { v[q] = __expf(v[q] - mx); s += v[q]; }
  float inv = 1.f / s;
#pragma unroll
  for (int q = 0; q < 9; ++q) dst[q] = f2bf(v[q] * inv);
}

// ---------------- K3: 9 shifted GEMMs  V[p] = shift_p(xb) @ w_v[p]^T -------
// 16x32 tile per wave; A rows gathered from shifted pixel, zero if OOB.
// Control re-converges before the WMMAs, so EXEC is all-1s at the matrix ops.
__global__ void vgemm_shift(const __bf16* __restrict__ xb, const __bf16* __restrict__ wv,
                            __bf16* __restrict__ V) {
  int wave = (blockIdx.x * blockDim.x + threadIdx.x) >> 5;
  int lane = threadIdx.x & 31;
  int tp  = wave & 3;            // 128/32 = 4 column-pair tiles
  int tmp = wave >> 2;
  int tm  = tmp & 4095;          // 65536/16 = 4096 row tiles
  int p   = tmp >> 12;           // 0..8
  int di = p / 3, dj = p % 3;
  int row = lane & 15, hi = lane >> 4;

  int m = tm * 16 + row;
  int b = m >> 14, r2 = m & 16383;
  int i = r2 >> 7, j = r2 & 127;
  int si = i + di - 1, sj = j + dj - 1;
  bool valid = (si >= 0) & (si < 128) & (sj >= 0) & (sj < 128);

  const __bf16* a  = xb + (size_t)((b << 14) + (si << 7) + sj) * C_DIM + hi * 8;
  const __bf16* b0 = wv + (size_t)p * C_DIM * C_DIM
                        + (size_t)(tp * 32 + row) * C_DIM + hi * 8;
  const __bf16* b1 = b0 + (size_t)16 * C_DIM;

  v16bf af[4], bf0[4], bf1[4];
#pragma unroll
  for (int kk = 0; kk < 4; ++kk) {
    if (valid) {
      af[kk] = load_frag(a + kk * 32);
    } else {
#pragma unroll
      for (int t = 0; t < 16; ++t) af[kk][t] = f2bf(0.0f);
    }
  }
#pragma unroll
  for (int kk = 0; kk < 4; ++kk) bf0[kk] = load_frag(b0 + kk * 32);
#pragma unroll
  for (int kk = 0; kk < 4; ++kk) bf1[kk] = load_frag(b1 + kk * 32);

  v8f acc0 = {}, acc1 = {};
#pragma unroll
  for (int kk = 0; kk < 4; ++kk) {
    acc0 = __builtin_amdgcn_wmma_f32_16x16x32_bf16(false, af[kk], false, bf0[kk],
                                                   (short)0, acc0, false, false);
    acc1 = __builtin_amdgcn_wmma_f32_16x16x32_bf16(false, af[kk], false, bf1[kk],
                                                   (short)0, acc1, false, false);
  }

  __bf16* d = V + (size_t)p * ((size_t)M_TOT * C_DIM)
                + (size_t)(tm * 16 + hi * 8) * C_DIM + tp * 32 + row;
#pragma unroll
  for (int r = 0; r < 8; ++r) {
    d[(size_t)r * C_DIM]      = f2bf(acc0[r]);
    d[(size_t)r * C_DIM + 16] = f2bf(acc1[r]);
  }
}

// ---------------- K4: attention apply + fold (gather form) ----------------
// y[b,i,j,c] = sum_{p=(di,dj) valid} sum_q att[b,h(c),n_p,p,q] * V[q][n_p,c]
// n_p = (i+1-di, j+1-dj).  Block = one pixel (128 threads, c = tid).
// The 9x re-read of V (151 MB) is spatially local and fits in the 192 MB L2.
__global__ void attn_fold(const __bf16* __restrict__ attb, const __bf16* __restrict__ V,
                          __bf16* __restrict__ yb) {
  int m = blockIdx.x;
  int c = threadIdx.x;
  int b = m >> 14, r2 = m & 16383;
  int i = r2 >> 7, j = r2 & 127;
  int h = c >> 5;
  float acc = 0.f;
#pragma unroll
  for (int di = 0; di < 3; ++di) {
    int u = i + 1 - di;
    if (u < 0 || u >= 128) continue;
#pragma unroll
    for (int dj = 0; dj < 3; ++dj) {
      int w = j + 1 - dj;
      if (w < 0 || w >= 128) continue;
      int p = di * 3 + dj;
      size_t nm = (size_t)((b << 14) + (u << 7) + w);
      const __bf16* arow = attb + nm * NATT + h * 81 + p * 9;
#pragma unroll
      for (int q = 0; q < 9; ++q) {
        float av = bf2f(arow[q]);
        float vv = bf2f(V[(size_t)q * ((size_t)M_TOT * C_DIM) + nm * C_DIM + c]);
        acc = fmaf(av, vv, acc);
      }
    }
  }
  yb[(size_t)m * C_DIM + c] = f2bf(acc);
}

// ---------------------------------------------------------------------------
extern "C" void kernel_launch(void* const* d_in, const int* in_sizes, int n_in,
                              void* d_out, int out_size, void* d_ws, size_t ws_size,
                              hipStream_t stream) {
  const float* x      = (const float*)d_in[0];   // [4,128,128,128]
  const float* w_qkv  = (const float*)d_in[1];   // [324,128]
  const float* w_v    = (const float*)d_in[2];   // [9,128,128]
  const float* w_proj = (const float*)d_in[3];   // [128,128]
  float* out = (float*)d_out;                    // [4,128,128,128]

  // -- workspace layout (256B aligned) --
  char* ws = (char*)d_ws;
  size_t off = 0;
  auto take = [&](size_t bytes) { char* p = ws + off; off += (bytes + 255) & ~(size_t)255; return p; };
  __bf16* xb      = (__bf16*)take((size_t)M_TOT * C_DIM * 2);            // 16.8 MB
  __bf16* wq      = (__bf16*)take((size_t)NPAD * C_DIM * 2);             // 86 KB
  __bf16* wvb     = (__bf16*)take((size_t)9 * C_DIM * C_DIM * 2);        // 288 KB
  __bf16* wpb     = (__bf16*)take((size_t)C_DIM * C_DIM * 2);            // 32 KB
  float*  att_raw = (float*) take((size_t)M_TOT * NPAD * 4);             // 88 MB
  __bf16* attb    = (__bf16*)take((size_t)M_TOT * NATT * 2);             // 42 MB
  __bf16* V       = (__bf16*)take((size_t)9 * M_TOT * C_DIM * 2);        // 151 MB
  __bf16* yb      = (__bf16*)take((size_t)M_TOT * C_DIM * 2);            // 16.8 MB
  (void)ws_size;

  // K0: conversions
  {
    int n = M_TOT * C_DIM;
    cvt_f32_bf16<<<(n + 255) / 256, 256, 0, stream>>>(x, xb, n);
    pack_wq<<<(NPAD * C_DIM + 255) / 256, 256, 0, stream>>>(w_qkv, wq);
    n = 9 * C_DIM * C_DIM;
    cvt_f32_bf16<<<(n + 255) / 256, 256, 0, stream>>>(w_v, wvb, n);
    n = C_DIM * C_DIM;
    cvt_f32_bf16<<<(n + 255) / 256, 256, 0, stream>>>(w_proj, wpb, n);
  }

  // K1: att_raw = xb @ wq^T   (tiles: 4096 x 21 waves)
  {
    int waves = (M_TOT / 16) * (NPAD / 16);
    gemm_bf16_f32<<<waves / 8, 256, 0, stream>>>(xb, wq, att_raw, NPAD / 16, NPAD);
  }

  // K2: softmax
  {
    int n = M_TOT * 36;
    softmax9<<<(n + 255) / 256, 256, 0, stream>>>(att_raw, attb);
  }

  // K3: V[p] = shift_p(xb) @ w_v[p]^T   (9 * 4096 * 4 waves, 16x32 tiles)
  {
    int waves = 9 * (M_TOT / 16) * (C_DIM / 32);
    vgemm_shift<<<waves / 8, 256, 0, stream>>>(xb, wvb, V);
  }

  // K4: attention apply + fold
  attn_fold<<<M_TOT, C_DIM, 0, stream>>>(attb, V, yb);

  // K5: out = yb @ w_proj^T   (4096 * 4 waves, 16x32 tiles)
  {
    int waves = (M_TOT / 16) * (C_DIM / 32);
    gemm_bf16_f32_x2<<<waves / 8, 256, 0, stream>>>(yb, wpb, out);
  }
}
